// BailingMoeV2_5LinearAttention_64922725646345
// MI455X (gfx1250) — compile-verified, matrix operand order checked by reading, not run on hardware
//
#include <hip/hip_runtime.h>

// ---------------------------------------------------------------------------
// Types for CDNA5 WMMA
// ---------------------------------------------------------------------------
typedef __attribute__((ext_vector_type(16))) __bf16 v16bf;
typedef __attribute__((ext_vector_type(8)))  __bf16 v8bf;
typedef __attribute__((ext_vector_type(8)))  float  v8f;

#define B_    2
#define T_    4096
#define HID_  2048
#define H_    16
#define D_    128
#define NQKV  (3 * H_ * D_)   // 6144
#define CHUNK_ 64

// ---------------------------------------------------------------------------
// Helpers
// ---------------------------------------------------------------------------
__device__ __forceinline__ v8f wmma_bf16(v16bf a, v16bf b, v8f c) {
  // (neg_a, A, neg_b, B, c_mod, C, reuse_a, reuse_b)
  return __builtin_amdgcn_wmma_f32_16x16x32_bf16(false, a, false, b, (short)0, c,
                                                 false, false);
}

// Load one lane's 16-element bf16 fragment for the 16x16x32 op.
// Per ISA layout: lo half-wave holds K[0:8) and K[16:24), hi half-wave holds
// K[8:16) and K[24:32) of its row. 16B-aligned LDS vector loads.
__device__ __forceinline__ v16bf ldfrag(const __bf16* rowp, int half) {
  const v8bf lo = *reinterpret_cast<const v8bf*>(rowp + half * 8);
  const v8bf hi = *reinterpret_cast<const v8bf*>(rowp + 16 + half * 8);
  return __builtin_shufflevector(lo, hi, 0, 1, 2, 3, 4, 5, 6, 7,
                                 8, 9, 10, 11, 12, 13, 14, 15);
}

__device__ __forceinline__ float head_slope(int h) {
  // alibi: start = 2^(-2^-(log2(16)-3)) = 2^-0.5 ; slope_h = start^(h+1)
  const float scale = 1.0f - (12.0f - 1.0f) / (32.0f - 1.0f) + 1e-5f;
  return -exp2f(-0.5f * (float)(h + 1)) * scale;
}

// ---------------------------------------------------------------------------
// fp32 -> bf16 conversion
// ---------------------------------------------------------------------------
__global__ __launch_bounds__(256) void f2bf_kernel(const float* __restrict__ in,
                                                   __bf16* __restrict__ out,
                                                   long long n) {
  long long i = (long long)blockIdx.x * blockDim.x + threadIdx.x;
  long long stride = (long long)gridDim.x * blockDim.x;
  for (; i < n; i += stride) out[i] = (__bf16)in[i];
}

// ---------------------------------------------------------------------------
// Generic GEMM: C[M,N] (fp32) = A[M,K] (bf16, row-major) * W[N,K]^T (bf16)
// 128x128 tile per 256-thread block (8 waves), K staged 32 at a time in LDS.
// ---------------------------------------------------------------------------
__global__ __launch_bounds__(256) void gemm_bf16(const __bf16* __restrict__ A,
                                                 const __bf16* __restrict__ W,
                                                 float* __restrict__ C,
                                                 int M, int N, int K) {
  __shared__ __align__(16) __bf16 As[128][40];
  __shared__ __align__(16) __bf16 Ws[128][40];

  const int tid  = threadIdx.x;
  const int lane = tid & 31;
  const int wid  = tid >> 5;
  const int r    = lane & 15;
  const int half = (lane >> 4) & 1;
  const int wm   = wid >> 1;       // 0..3 -> 32-row strips
  const int wn   = wid & 1;        // 0..1 -> 64-col strips

  const int m0 = blockIdx.y * 128;
  const int n0 = blockIdx.x * 128;

  v8f acc[2][4];
#pragma unroll
  for (int mi = 0; mi < 2; ++mi)
#pragma unroll
    for (int ni = 0; ni < 4; ++ni) acc[mi][ni] = (v8f)0.0f;

  const int lrow = tid >> 1;       // 0..127
  const int lseg = tid & 1;        // which 16-elem half of the 32-wide slab

  for (int k0 = 0; k0 < K; k0 += 32) {
    __syncthreads();
    {
      const __bf16* ga = A + (size_t)(m0 + lrow) * K + k0 + lseg * 16;
      const __bf16* gw = W + (size_t)(n0 + lrow) * K + k0 + lseg * 16;
      uint4 a0 = *(const uint4*)(ga);
      uint4 a1 = *(const uint4*)(ga + 8);
      uint4 w0 = *(const uint4*)(gw);
      uint4 w1 = *(const uint4*)(gw + 8);
      *(uint4*)&As[lrow][lseg * 16]     = a0;
      *(uint4*)&As[lrow][lseg * 16 + 8] = a1;
      *(uint4*)&Ws[lrow][lseg * 16]     = w0;
      *(uint4*)&Ws[lrow][lseg * 16 + 8] = w1;
    }
    __syncthreads();

    v16bf af[2], bf[4];
#pragma unroll
    for (int mi = 0; mi < 2; ++mi)
      af[mi] = ldfrag(&As[wm * 32 + mi * 16 + r][0], half);
#pragma unroll
    for (int ni = 0; ni < 4; ++ni)
      bf[ni] = ldfrag(&Ws[wn * 64 + ni * 16 + r][0], half);
#pragma unroll
    for (int mi = 0; mi < 2; ++mi)
#pragma unroll
      for (int ni = 0; ni < 4; ++ni)
        acc[mi][ni] = wmma_bf16(af[mi], bf[ni], acc[mi][ni]);
  }

  // C/D layout: VGPR i -> M = i + half*8 ; N = lane&15
#pragma unroll
  for (int mi = 0; mi < 2; ++mi) {
#pragma unroll
    for (int ni = 0; ni < 4; ++ni) {
      const int col = n0 + wn * 64 + ni * 16 + r;
#pragma unroll
      for (int i = 0; i < 8; ++i) {
        const int row = m0 + wm * 32 + mi * 16 + half * 8 + i;
        C[(size_t)row * N + col] = acc[mi][ni][i];
      }
    }
  }
}

// ---------------------------------------------------------------------------
// Per-row RMSNorm + RoPE (in place on the qkv buffer), one 128-thread block
// per (b, t, h, {q|k}) row.
// ---------------------------------------------------------------------------
__global__ __launch_bounds__(128) void qk_norm_rope(float* __restrict__ qkv,
                                                    const float* __restrict__ qw,
                                                    const float* __restrict__ kw,
                                                    const int* __restrict__ pos_ids) {
  int idx = blockIdx.x;
  const int which = idx & 1;  idx >>= 1;
  const int h = idx & (H_ - 1);  idx >>= 4;
  const int t = idx & (T_ - 1);  idx >>= 12;
  const int b = idx;

  float* row = qkv + ((size_t)(b * T_ + t)) * NQKV + which * (H_ * D_) + h * D_;
  const float* w = which ? kw : qw;
  const int d = threadIdx.x;

  float x = row[d];
  float s = x * x;
#pragma unroll
  for (int o = 16; o > 0; o >>= 1) s += __shfl_xor(s, o, 32);

  __shared__ float partial[4];
  __shared__ float ynorm[128];
  if ((threadIdx.x & 31) == 0) partial[threadIdx.x >> 5] = s;
  __syncthreads();
  const float var = (partial[0] + partial[1] + partial[2] + partial[3]) * (1.0f / 128.0f);
  float y = x * rsqrtf(var + 1e-6f) * w[d];
  ynorm[d] = y;
  __syncthreads();

  float out = y;
  if (d < 64) {
    const int j = d & 31;
    const float invf = __powf(10000.0f, -(float)j * (1.0f / 32.0f));
    const float p = (float)pos_ids[b * T_ + t];
    const float ang = p * invf;
    const float other = (d < 32) ? -ynorm[d + 32] : ynorm[d - 32];
    out = y * __cosf(ang) + other * __sinf(ang);
  }
  row[d] = out;
}

// ---------------------------------------------------------------------------
// Chunked GLA scan. One block per (b, h); state S kept as U[d'][d] = S[d][d']
// in LDS (fp32 master + bf16 operand copy). All matmuls via bf16 WMMA.
// ---------------------------------------------------------------------------
#define QS  136   // row stride for 128-wide bf16 tiles (padded)
#define KTS 72    // row stride for 64-wide bf16 tiles (padded)
#define ATS 72
#define US  136
#define GLA_LDS_BYTES (128*136*4 /*Uf*/ + 128*136*2 /*Ubf*/ + 3*(64*136*2) /*qs,qds,ks*/ \
                       + 2*(128*72*2) /*ktd,vt*/ + 64*72*2 /*att*/)

__global__ __launch_bounds__(256) void gla_scan(const float* __restrict__ qkv,
                                                float* __restrict__ o) {
  const int b = blockIdx.x >> 4;
  const int h = blockIdx.x & (H_ - 1);

  extern __shared__ char smem[];
  float*  Uf  = (float*)smem;                            // [128][136] f32
  __bf16* Ubf = (__bf16*)(smem + 128 * 136 * 4);         // [128][136]
  __bf16* qs  = Ubf + 128 * 136;                         // [64][136]
  __bf16* qds = qs  + 64 * 136;                          // [64][136]
  __bf16* ks  = qds + 64 * 136;                          // [64][136]
  __bf16* ktd = ks  + 64 * 136;                          // [128][72]
  __bf16* vt  = ktd + 128 * 72;                          // [128][72]
  __bf16* att = vt  + 128 * 72;                          // [64][72]

  const int tid  = threadIdx.x;
  const int lane = tid & 31;
  const int wid  = tid >> 5;
  const int r    = lane & 15;
  const int half = (lane >> 4) & 1;

  const float g     = head_slope(h);
  const float chdec = __expf(g * 64.0f);
  const float invsq = 0.08838834764831845f;  // D^-0.5

  for (int i = tid; i < 128 * 136; i += 256) { Uf[i] = 0.0f; Ubf[i] = (__bf16)0.0f; }
  __syncthreads();

  for (int ch = 0; ch < T_ / CHUNK_; ++ch) {
    const int t0 = ch * CHUNK_;

    // ---- stage q/k/v chunk into LDS (bf16, incl. transposed/decayed copies)
    {
      const int c  = tid >> 2;
      const int j0 = (tid & 3) * 32;
      const float qdec = __expf(g * ((float)c + 1.0f));
      const float kdec = __expf(g * (63.0f - (float)c));
      const float* qrow = qkv + ((size_t)(b * T_ + t0 + c)) * NQKV + h * D_;
      const float4* q4 = (const float4*)(qrow + j0);
      const float4* k4 = (const float4*)(qrow + 2048 + j0);
      const float4* v4 = (const float4*)(qrow + 4096 + j0);
#pragma unroll
      for (int jj = 0; jj < 8; ++jj) {
        const float4 qv = q4[jj], kv = k4[jj], vv = v4[jj];
        const float qa[4] = {qv.x, qv.y, qv.z, qv.w};
        const float ka[4] = {kv.x, kv.y, kv.z, kv.w};
        const float va[4] = {vv.x, vv.y, vv.z, vv.w};
#pragma unroll
        for (int m = 0; m < 4; ++m) {
          const int j = j0 + jj * 4 + m;
          const float q = qa[m] * invsq;
          qs [c * QS + j]  = (__bf16)q;
          qds[c * QS + j]  = (__bf16)(q * qdec);
          ks [c * QS + j]  = (__bf16)ka[m];
          ktd[j * KTS + c] = (__bf16)(ka[m] * kdec);
          vt [j * KTS + c] = (__bf16)va[m];
        }
      }
    }
    __syncthreads();

    // ---- phase A: att[c][e] = (q . k) with causal decay mask, -> bf16 LDS
    {
      const int mi = wid >> 1;
      const int ns = (wid & 1) * 2;
      v8f a0 = (v8f)0.0f, a1 = (v8f)0.0f;
#pragma unroll
      for (int kk = 0; kk < 4; ++kk) {
        const v16bf af = ldfrag(qs + (mi * 16 + r) * QS + kk * 32, half);
        const v16bf b0 = ldfrag(ks + (ns * 16 + r) * QS + kk * 32, half);
        const v16bf b1 = ldfrag(ks + ((ns + 1) * 16 + r) * QS + kk * 32, half);
        a0 = wmma_bf16(af, b0, a0);
        a1 = wmma_bf16(af, b1, a1);
      }
#pragma unroll
      for (int i = 0; i < 8; ++i) {
        const int c  = mi * 16 + half * 8 + i;
        const int e0 = ns * 16 + r;
        const int e1 = (ns + 1) * 16 + r;
        const float m0v = (c >= e0) ? a0[i] * __expf(g * (float)(c - e0)) : 0.0f;
        const float m1v = (c >= e1) ? a1[i] * __expf(g * (float)(c - e1)) : 0.0f;
        att[c * ATS + e0] = (__bf16)m0v;
        att[c * ATS + e1] = (__bf16)m1v;
      }
    }
    __syncthreads();

    // ---- phase B: o_chunk = att @ v  +  (q*qdec) @ S_old
    {
      const int mi = wid >> 1;
      const int nb = (wid & 1) * 4;
      v8f acc[4];
#pragma unroll
      for (int t = 0; t < 4; ++t) acc[t] = (v8f)0.0f;

#pragma unroll
      for (int kk = 0; kk < 2; ++kk) {            // intra: K = 64 (e)
        const v16bf af = ldfrag(att + (mi * 16 + r) * ATS + kk * 32, half);
#pragma unroll
        for (int t = 0; t < 4; ++t) {
          const v16bf bf = ldfrag(vt + ((nb + t) * 16 + r) * KTS + kk * 32, half);
          acc[t] = wmma_bf16(af, bf, acc[t]);
        }
      }
#pragma unroll
      for (int kk = 0; kk < 4; ++kk) {            // inter: K = 128 (d)
        const v16bf af = ldfrag(qds + (mi * 16 + r) * QS + kk * 32, half);
#pragma unroll
        for (int t = 0; t < 4; ++t) {
          const v16bf bf = ldfrag(Ubf + ((nb + t) * 16 + r) * US + kk * 32, half);
          acc[t] = wmma_bf16(af, bf, acc[t]);
        }
      }
#pragma unroll
      for (int t = 0; t < 4; ++t) {
        const int d = (nb + t) * 16 + r;
#pragma unroll
        for (int i = 0; i < 8; ++i) {
          const int c = mi * 16 + half * 8 + i;
          o[((size_t)(b * T_ + t0 + c)) * (H_ * D_) + h * D_ + d] = acc[t][i];
        }
      }
    }
    __syncthreads();

    // ---- phase C: U = chdec*U + v^T @ (k*kdec)   (wave w owns rows d' in [16w,16w+16))
    {
      const int mi = wid;
      v8f acc[8];
#pragma unroll
      for (int ni = 0; ni < 8; ++ni) {
#pragma unroll
        for (int i = 0; i < 8; ++i)
          acc[ni][i] = chdec * Uf[(mi * 16 + half * 8 + i) * US + ni * 16 + r];
      }
#pragma unroll
      for (int kk = 0; kk < 2; ++kk) {            // K = 64 (e)
        const v16bf af = ldfrag(vt + (mi * 16 + r) * KTS + kk * 32, half);
#pragma unroll
        for (int ni = 0; ni < 8; ++ni) {
          const v16bf bf = ldfrag(ktd + (ni * 16 + r) * KTS + kk * 32, half);
          acc[ni] = wmma_bf16(af, bf, acc[ni]);
        }
      }
#pragma unroll
      for (int ni = 0; ni < 8; ++ni) {
#pragma unroll
        for (int i = 0; i < 8; ++i) {
          const int row = mi * 16 + half * 8 + i;
          const int col = ni * 16 + r;
          Uf [row * US + col] = acc[ni][i];
          Ubf[row * US + col] = (__bf16)acc[ni][i];
        }
      }
    }
    __syncthreads();
  }
}

// ---------------------------------------------------------------------------
// Group RMSNorm (groups of 128) * sigmoid(gate), emit bf16 for final GEMM
// ---------------------------------------------------------------------------
__global__ __launch_bounds__(128) void gate_norm(const float* __restrict__ o,
                                                 const float* __restrict__ gate,
                                                 const float* __restrict__ gw,
                                                 __bf16* __restrict__ out) {
  const int idx = blockIdx.x;            // b*T*H + ...
  const int h  = idx & (H_ - 1);
  const int bt = idx >> 4;
  const size_t base = (size_t)bt * (H_ * D_) + h * D_;
  const int d = threadIdx.x;

  const float x = o[base + d];
  float s = x * x;
#pragma unroll
  for (int oo = 16; oo > 0; oo >>= 1) s += __shfl_xor(s, oo, 32);
  __shared__ float partial[4];
  if ((threadIdx.x & 31) == 0) partial[threadIdx.x >> 5] = s;
  __syncthreads();
  const float var = (partial[0] + partial[1] + partial[2] + partial[3]) * (1.0f / 128.0f);

  float y = x * rsqrtf(var + 1e-6f) * gw[h * D_ + d];
  const float gl = gate[base + d];
  y *= 1.0f / (1.0f + __expf(-gl));
  out[base + d] = (__bf16)y;
}

// ---------------------------------------------------------------------------
// Launch
// ---------------------------------------------------------------------------
extern "C" void kernel_launch(void* const* d_in, const int* in_sizes, int n_in,
                              void* d_out, int out_size, void* d_ws, size_t ws_size,
                              hipStream_t stream) {
  (void)in_sizes; (void)n_in; (void)out_size; (void)ws_size;
  const float* hs      = (const float*)d_in[0];
  const float* w_qkv   = (const float*)d_in[1];
  const float* q_ln_w  = (const float*)d_in[2];
  const float* k_ln_w  = (const float*)d_in[3];
  const float* g_nw    = (const float*)d_in[4];
  const float* w_g     = (const float*)d_in[5];
  const float* w_dense = (const float*)d_in[6];
  const int*   pos     = (const int*)d_in[7];
  float* out = (float*)d_out;

  char* ws = (char*)d_ws;
  auto take = [&](size_t bytes) -> char* {
    char* p = ws;
    ws += (bytes + 255) & ~(size_t)255;
    return p;
  };
  __bf16* hs_bf   = (__bf16*)take((size_t)B_ * T_ * HID_ * 2);
  __bf16* wqkv_bf = (__bf16*)take((size_t)NQKV * HID_ * 2);
  __bf16* wg_bf   = (__bf16*)take((size_t)HID_ * HID_ * 2);
  __bf16* wd_bf   = (__bf16*)take((size_t)HID_ * HID_ * 2);
  float*  qkv_f   = (float*)take((size_t)B_ * T_ * NQKV * 4);
  float*  gate_f  = (float*)take((size_t)B_ * T_ * HID_ * 4);
  float*  o_f     = (float*)take((size_t)B_ * T_ * HID_ * 4);
  __bf16* o_bf    = (__bf16*)take((size_t)B_ * T_ * HID_ * 2);

  f2bf_kernel<<<2048, 256, 0, stream>>>(hs, hs_bf, (long long)B_ * T_ * HID_);
  f2bf_kernel<<<2048, 256, 0, stream>>>(w_qkv, wqkv_bf, (long long)NQKV * HID_);
  f2bf_kernel<<<1024, 256, 0, stream>>>(w_g, wg_bf, (long long)HID_ * HID_);
  f2bf_kernel<<<1024, 256, 0, stream>>>(w_dense, wd_bf, (long long)HID_ * HID_);

  gemm_bf16<<<dim3(NQKV / 128, (B_ * T_) / 128), 256, 0, stream>>>(
      hs_bf, wqkv_bf, qkv_f, B_ * T_, NQKV, HID_);
  gemm_bf16<<<dim3(HID_ / 128, (B_ * T_) / 128), 256, 0, stream>>>(
      hs_bf, wg_bf, gate_f, B_ * T_, HID_, HID_);

  qk_norm_rope<<<B_ * T_ * H_ * 2, 128, 0, stream>>>(qkv_f, q_ln_w, k_ln_w, pos);

  gla_scan<<<B_ * H_, 256, GLA_LDS_BYTES, stream>>>(qkv_f, o_f);

  gate_norm<<<B_ * T_ * H_, 128, 0, stream>>>(o_f, gate_f, g_nw, o_bf);

  gemm_bf16<<<dim3(HID_ / 128, (B_ * T_) / 128), 256, 0, stream>>>(
      o_bf, wd_bf, out, B_ * T_, HID_, HID_);
}
